// RecurrentSpatialQNetwork_34909494182358
// MI455X (gfx1250) — matrix operand order repaired and checked by reading, not compiled, not run on hardware
//
#include <hip/hip_runtime.h>

// ---------------- CDNA5 WMMA types (FP16 pipeline) ----------------
typedef __attribute__((ext_vector_type(16))) _Float16 v16h;
typedef __attribute__((ext_vector_type(2)))  __fp16   v2fp;
typedef __attribute__((ext_vector_type(8)))  float    v8f;

union HVec { uint4 q[2]; v16h v; };

// ---------------- workspace layout (bytes) ----------------
#define OFF_WC1   0u
#define OFF_WC2   25600u
#define OFF_WVIS  691200u
#define OFF_WSM   904192u
#define OFF_WLSTM 910336u
#define OFF_WQ1   1893376u
#define OFF_WQ2   1958912u
#define OFF_B1    1963008u
#define OFF_B2    1964608u
#define OFF_BSM   1967808u
#define OFF_BG    1968192u
#define OFF_BQ2   1972288u

// swizzled-weight element counts (f16 elems), cumulative bounds used in prep
#define CUM_WC1   12800
#define CUM_WC2   345600
#define CUM_WVIS  452096
#define CUM_WSM   455168
#define CUM_WLSTM 946688
#define CUM_WQ1   979456
#define CUM_WQ2   981504
#define PREP_TOTAL (CUM_WQ2 + 2336)

// ---------------- LDS layout (bytes) ----------------
#define S_A 0          // 56KB : A2 [32][832] f16  /  comb [32][480] f16
#define S_B 57344      // 32KB : A1 [32][416] f16  /  visout [32][128] f32 / hbuf [32][256] f32
#define S_C 90112      //  4KB : A0 [32][32] f16 + Asm [32][32] f16
#define S_D 94208      // 16KB : Aq1 [32][256] f16 /  Aq2 [32][128] f16
#define S_E 110592     // 16KB : q1out [32][128] f32
#define SMEM_BYTES 126976

// f32 -> f16 bit pattern (v_cvt_f16_f32)
__device__ __forceinline__ unsigned short f2h(float f) {
  union { _Float16 h; unsigned short s; } u; u.h = (_Float16)f; return u.s;
}
// packed f32x2 -> f16x2 (single v_cvt_pk_rtz_f16_f32)
__device__ __forceinline__ unsigned pack_h2(float a, float b) {
  union { v2fp v; unsigned u; } c;
  c.v = __builtin_amdgcn_cvt_pkrtz(a, b);
  return c.u;
}
#if __has_builtin(__builtin_amdgcn_tanhf)
__device__ __forceinline__ float fast_tanh(float x) { return __builtin_amdgcn_tanhf(x); }
#elif __has_builtin(__builtin_amdgcn_tanh_f32)
__device__ __forceinline__ float fast_tanh(float x) { return __builtin_amdgcn_tanh_f32(x); }
#else
__device__ __forceinline__ float fast_tanh(float x) { return tanhf(x); }
#endif
__device__ __forceinline__ float sigm(float x) { return 0.5f + 0.5f * fast_tanh(0.5f * x); }

// ---------------- prep kernel: build f16 swizzled weights + fused biases ----------------
__global__ __launch_bounds__(256) void prep_weights(
    const float* __restrict__ conv1_w, const float* __restrict__ conv1_b,
    const float* __restrict__ conv2_w, const float* __restrict__ conv2_b,
    const float* __restrict__ vis_w,
    const float* __restrict__ pos_w, const float* __restrict__ pos_b,
    const float* __restrict__ met_w, const float* __restrict__ met_b,
    const float* __restrict__ aff_w, const float* __restrict__ aff_b,
    const float* __restrict__ w_ih, const float* __restrict__ w_hh,
    const float* __restrict__ b_ih, const float* __restrict__ b_hh,
    const float* __restrict__ q1_w, const float* __restrict__ q2_w,
    const float* __restrict__ q2_b, char* __restrict__ ws)
{
  int idx = blockIdx.x * blockDim.x + threadIdx.x;
  if (idx < CUM_WQ2) {
    int region, base, KT; unsigned off;
    if      (idx < CUM_WC1 ) { region=0; base=0;         KT=1;  off=OFF_WC1; }
    else if (idx < CUM_WC2 ) { region=1; base=CUM_WC1;   KT=13; off=OFF_WC2; }
    else if (idx < CUM_WVIS) { region=2; base=CUM_WC2;   KT=26; off=OFF_WVIS; }
    else if (idx < CUM_WSM ) { region=3; base=CUM_WVIS;  KT=1;  off=OFF_WSM; }
    else if (idx < CUM_WLSTM){ region=4; base=CUM_WSM;   KT=15; off=OFF_WLSTM; }
    else if (idx < CUM_WQ1 ) { region=5; base=CUM_WLSTM; KT=8;  off=OFF_WQ1; }
    else                     { region=6; base=CUM_WQ1;   KT=4;  off=OFF_WQ2; }
    int li   = idx - base;
    int tile = li >> 9, rr = li & 511, lane = rr >> 4, e = rr & 15;
    int kt = tile % KT, nt = tile / KT;
    // B-operand register image: lane L holds (K = kt*32 + (L/16)*8 + (e%8) + (e/8)*16, N = nt*16 + (L%16))
    int K = kt * 32 + ((lane >> 4) << 3) + (e & 7) + ((e >> 3) << 4);
    int N = nt * 16 + (lane & 15);
    float v = 0.f;
    switch (region) {
      case 0: { // conv1 unrolled: K=input pos (<25), N = c*25 + p (<400)
        if (K < 25) {
          int c = N / 25, p = N % 25, pr = p / 5, pc = p % 5, qr = K / 5, qc = K % 5;
          int dr = qr - pr + 1, dc = qc - pc + 1;
          if (dr >= 0 && dr < 3 && dc >= 0 && dc < 3) v = conv1_w[c * 9 + dr * 3 + dc];
        }
      } break;
      case 1: { // conv2 unrolled: K = ci*25 + q (<400), N = co*25 + p (<800)
        if (K < 400) {
          int ci = K / 25, q = K % 25, co = N / 25, p = N % 25;
          int pr = p / 5, pc = p % 5, qr = q / 5, qc = q % 5;
          int dr = qr - pr + 1, dc = qc - pc + 1;
          if (dr >= 0 && dr < 3 && dc >= 0 && dc < 3) v = conv2_w[(co * 16 + ci) * 9 + dr * 3 + dc];
        }
      } break;
      case 2: { if (K < 800) v = vis_w[N * 800 + K]; } break;          // vis_w (128,800)
      case 3: {                                                        // [pos|met|aff] -> 96 cols
        if (N < 32)      { if (K < 2)             v = pos_w[N * 2 + K]; }
        else if (N < 64) { if (K >= 2 && K < 10)  v = met_w[(N - 32) * 8 + (K - 2)]; }
        else             { if (K >= 10 && K < 25) v = aff_w[(N - 64) * 15 + (K - 10)]; }
      } break;
      case 4: { v = (K < 224) ? w_ih[N * 224 + K] : w_hh[N * 256 + (K - 224)]; } break;
      case 5: { v = q1_w[N * 256 + K]; } break;
      case 6: { if (N < 8) v = q2_w[N * 128 + K]; } break;
    }
    *((unsigned short*)(ws + off) + (size_t)tile * 512 + lane * 16 + e) = f2h(v);
  } else if (idx < PREP_TOTAL) {
    int bi = idx - CUM_WQ2;
    if      (bi < 400)  ((float*)(ws + OFF_B1))[bi] = conv1_b[bi / 25];
    else if (bi < 1200) { int n = bi - 400;  ((float*)(ws + OFF_B2))[n] = conv2_b[n / 25]; }
    else if (bi < 1296) { int n = bi - 1200;
      ((float*)(ws + OFF_BSM))[n] = (n < 32) ? pos_b[n] : (n < 64) ? met_b[n - 32] : aff_b[n - 64]; }
    else if (bi < 2320) { int n = bi - 1296; ((float*)(ws + OFF_BG))[n] = b_ih[n] + b_hh[n]; }
    else                { int n = bi - 2320; ((float*)(ws + OFF_BQ2))[n] = (n < 8) ? q2_b[n] : 0.f; }
  }
}

// ---------------- WMMA helpers ----------------
__device__ __forceinline__ v8f wmma_acc(const HVec& A, const HVec& B, v8f c) {
  return __builtin_amdgcn_wmma_f32_16x16x32_f16(false, A.v, false, B.v, (short)0, c, false, false);
}
__device__ __forceinline__ void ld2(const void* p0, const void* p1, HVec& V) {
  V.q[0] = *(const uint4*)p0;
  V.q[1] = *(const uint4*)p1;
}
// software-pipelined K loop with hoisted base pointers; loop offsets are
// compile-time constants that fold into DS/global immediate offset fields.
__device__ __forceinline__ v8f gemm_acc(const unsigned short* Ast, int ld, const char* W,
                                        int KT, int msub, int nt, int lane) {
  // A: lane L (M = L&15) holds K = kt*32 + {half*8..+7, 16+half*8..+7}, half = L>>4
  const unsigned short* Ab = Ast + (msub * 16 + (lane & 15)) * ld + ((lane >> 4) << 3);
  const char*           Wb = W + ((size_t)(nt * KT) << 10) + (lane << 5);
  v8f acc = {};
  HVec A, B;
  ld2(Ab, Ab + 16, A);
  ld2(Wb, Wb + 16, B);
#pragma unroll
  for (int kt = 0; kt < KT - 1; ++kt) {
    HVec An, Bn;
    ld2(Ab + (kt + 1) * 32, Ab + (kt + 1) * 32 + 16, An);
    ld2(Wb + ((kt + 1) << 10), Wb + ((kt + 1) << 10) + 16, Bn);
    acc = wmma_acc(A, B, acc);
    A = An; B = Bn;
  }
  return wmma_acc(A, B, acc);
}
// D-register epilogue: +bias, relu, f32x2 -> f16x2 packed convert, 2x ds_store_b16
__device__ __forceinline__ void store_tile_f16_relu(v8f acc, float bv, unsigned short* dst,
                                                    int ld, int n, int mbase) {
#pragma unroll
  for (int r = 0; r < 8; r += 2) {
    float v0 = fmaxf(acc[r]     + bv, 0.f);
    float v1 = fmaxf(acc[r + 1] + bv, 0.f);
    unsigned p = pack_h2(v0, v1);
    dst[(mbase + r)     * ld + n] = (unsigned short)p;
    dst[(mbase + r + 1) * ld + n] = (unsigned short)(p >> 16);
  }
}
// per-row LayerNorm over LDS f32 rows -> f16 staging (col 0..W-1); 8 threads / row.
// strided reduction read, contiguous packed-b32 writes.
__device__ __forceinline__ void ln_rows(const float* src, int W, const float* __restrict__ g,
                                        const float* __restrict__ b, unsigned short* dst,
                                        int ldD, bool relu, int tid) {
  int r = tid >> 3, sub = tid & 7;
  float s = 0.f, s2 = 0.f;
  for (int c = sub; c < W; c += 8) { float v = src[r * W + c]; s += v; s2 += v * v; }
#pragma unroll
  for (int m = 1; m < 8; m <<= 1) { s += __shfl_xor(s, m, 32); s2 += __shfl_xor(s2, m, 32); }
  float inv = 1.f / (float)W;
  float mean = s * inv;
  float rs = rsqrtf(s2 * inv - mean * mean + 1e-5f);
  int cw = W >> 3, c0 = sub * cw;
  for (int c = c0; c < c0 + cw; c += 2) {
    float v0 = (src[r * W + c]     - mean) * rs * g[c]     + b[c];
    float v1 = (src[r * W + c + 1] - mean) * rs * g[c + 1] + b[c + 1];
    if (relu) { v0 = fmaxf(v0, 0.f); v1 = fmaxf(v1, 0.f); }
    *(unsigned*)(dst + r * ldD + c) = pack_h2(v0, v1);
  }
}

// ---------------- fused network kernel: 32 batch rows / block ----------------
__global__ __launch_bounds__(256) void rsqn_fused_wmma(
    const float* __restrict__ obs, const float* __restrict__ h0, const float* __restrict__ c0,
    const float* __restrict__ vis_bv, const float* __restrict__ vis_g, const float* __restrict__ vis_beta,
    const float* __restrict__ ln_g, const float* __restrict__ ln_b,
    const float* __restrict__ q1_bv, const float* __restrict__ q1_g, const float* __restrict__ q1_beta,
    const char* __restrict__ ws, float* __restrict__ out, int Btot)
{
  extern __shared__ char smem[];
  unsigned short* A0   = (unsigned short*)(smem + S_C);
  unsigned short* Asm  = (unsigned short*)(smem + S_C + 2048);
  unsigned short* A1   = (unsigned short*)(smem + S_B);   // [32][416]
  unsigned short* A2   = (unsigned short*)(smem + S_A);   // [32][832]
  float*          viso = (float*)(smem + S_B);            // [32][128]
  unsigned short* comb = (unsigned short*)(smem + S_A);   // [32][480]
  float*          hbuf = (float*)(smem + S_B);            // [32][256]
  unsigned short* Aq1  = (unsigned short*)(smem + S_D);   // [32][256]
  float*          q1o  = (float*)(smem + S_E);            // [32][128]
  unsigned short* Aq2  = (unsigned short*)(smem + S_D);   // [32][128]

  const float* b1  = (const float*)(ws + OFF_B1);
  const float* b2  = (const float*)(ws + OFF_B2);
  const float* bsm = (const float*)(ws + OFF_BSM);
  const float* bg  = (const float*)(ws + OFF_BG);
  const float* bq2 = (const float*)(ws + OFF_BQ2);

  const int tid = threadIdx.x, lane = tid & 31;
  // wave index forced into an SGPR: unit loops compile to scalar branches,
  // keeping EXEC all-ones around WMMA (ISA requirement).
  const int wave = __builtin_amdgcn_readfirstlane(tid >> 5);
  const int Nl = lane & 15, half = lane >> 4;
  const int rowbase = blockIdx.x * 32;

  float* outQ = out;
  float* outH = out + (size_t)Btot * 8;
  float* outC = out + (size_t)Btot * (8 + 256);

  // ---- stage 1: stage obs grid + small features (f16), zero K-pad columns ----
  for (int idx = tid; idx < 32 * 64; idx += 256) {
    int r = idx >> 6, c = idx & 63;
    const float* orow = obs + (size_t)(rowbase + r) * 50;
    if (c < 32) A0 [r * 32 + c]        = f2h(c  < 25 ? orow[c]       : 0.f);
    else        { int c2 = c - 32; Asm[r * 32 + c2] = f2h(c2 < 25 ? orow[25 + c2] : 0.f); }
  }
  for (int idx = tid; idx < 32 * 16; idx += 256) { int r = idx >> 4; A1[r * 416 + 400 + (idx & 15)] = 0; }
  for (int idx = tid; idx < 32 * 32; idx += 256) { int r = idx >> 5; A2[r * 832 + 800 + (idx & 31)] = 0; }
  __syncthreads();

  // ---- stage 2: conv1 as dense GEMM (32x32 @ 32x400), bias+relu -> A1 ----
  for (int u = wave; u < 50; u += 8) {
    int msub = u / 25, nt = u % 25;
    v8f acc = gemm_acc(A0, 32, (const char*)ws + OFF_WC1, 1, msub, nt, lane);
    int n = nt * 16 + Nl;
    store_tile_f16_relu(acc, b1[n], A1, 416, n, msub * 16 + 8 * half);
  }
  __syncthreads();

  // ---- stage 3: conv2 as dense GEMM (32x416 @ 416x800), bias+relu -> A2 ----
  for (int u = wave; u < 100; u += 8) {
    int msub = u / 50, nt = u % 50;
    v8f acc = gemm_acc(A1, 416, (const char*)ws + OFF_WC2, 13, msub, nt, lane);
    int n = nt * 16 + Nl;
    store_tile_f16_relu(acc, b2[n], A2, 832, n, msub * 16 + 8 * half);
  }
  __syncthreads();

  // ---- stage 4: vision GEMM (32x832 @ 832x128) + bias -> f32 (LN next) ----
  for (int u = wave; u < 16; u += 8) {
    int msub = u >> 3, nt = u & 7;
    v8f acc = gemm_acc(A2, 832, (const char*)ws + OFF_WVIS, 26, msub, nt, lane);
    int n = nt * 16 + Nl; float bv = vis_bv[n];
#pragma unroll
    for (int r = 0; r < 8; ++r) viso[(msub * 16 + r + 8 * half) * 128 + n] = acc[r] + bv;
  }
  __syncthreads();

  // ---- stage 5: build combined[480] = LN(vis)+relu | small relu | h0(f16) ----
  ln_rows(viso, 128, vis_g, vis_beta, comb, 480, true, tid);
  {
    // 8 threads/row, 32 contiguous cols each -> packed b32 LDS stores
    int r = tid >> 3, k0 = (tid & 7) * 32;
    const float* srcp = h0 + (size_t)(rowbase + r) * 256 + k0;
    unsigned short* dstp = comb + r * 480 + 224 + k0;
#pragma unroll
    for (int k = 0; k < 32; k += 2)
      *(unsigned*)(dstp + k) = pack_h2(srcp[k], srcp[k + 1]);
  }
  for (int u = wave; u < 12; u += 8) {
    int msub = u / 6, nt = u % 6;
    v8f acc = gemm_acc(Asm, 32, (const char*)ws + OFF_WSM, 1, msub, nt, lane);
    int n = nt * 16 + Nl;
    store_tile_f16_relu(acc, bsm[n], comb + 128, 480, n, msub * 16 + 8 * half);
  }
  __syncthreads();

  // ---- stage 6: LSTM — 4 gate tiles share each A load; software-pipelined ----
  for (int u = wave; u < 32; u += 8) {
    int msub = u >> 4, j = u & 15;
    const char* WL = (const char*)ws + OFF_WLSTM;
    // hoisted per-operand base pointers; loop offsets are constants
    const unsigned short* Ab = comb + (msub * 16 + (lane & 15)) * 480 + ((lane >> 4) << 3);
    const char* W0 = WL + ((size_t)((0 * 16 + j) * 15) << 10) + (lane << 5);
    const char* W1 = WL + ((size_t)((1 * 16 + j) * 15) << 10) + (lane << 5);
    const char* W2 = WL + ((size_t)((2 * 16 + j) * 15) << 10) + (lane << 5);
    const char* W3 = WL + ((size_t)((3 * 16 + j) * 15) << 10) + (lane << 5);
    v8f ai = {}, af = {}, ag = {}, ao = {};
    HVec A, B0, B1, B2, B3;
    ld2(Ab, Ab + 16, A);
    ld2(W0, W0 + 16, B0);
    ld2(W1, W1 + 16, B1);
    ld2(W2, W2 + 16, B2);
    ld2(W3, W3 + 16, B3);
#pragma unroll
    for (int kt = 0; kt < 15; ++kt) {
      HVec An, C0, C1, C2, C3;
      if (kt + 1 < 15) {
        const int ao_ = (kt + 1) * 32;      // A element offset
        const int bo_ = (kt + 1) << 10;     // B byte offset
        ld2(Ab + ao_, Ab + ao_ + 16, An);
        ld2(W0 + bo_, W0 + bo_ + 16, C0);
        ld2(W1 + bo_, W1 + bo_ + 16, C1);
        ld2(W2 + bo_, W2 + bo_ + 16, C2);
        ld2(W3 + bo_, W3 + bo_ + 16, C3);
      }
      ai = wmma_acc(A, B0, ai);
      af = wmma_acc(A, B1, af);
      ag = wmma_acc(A, B2, ag);
      ao = wmma_acc(A, B3, ao);
      A = An; B0 = C0; B1 = C1; B2 = C2; B3 = C3;
    }
    int col = j * 16 + Nl;
    float bi_ = bg[col], bf_ = bg[256 + col], bg_ = bg[512 + col], bo_ = bg[768 + col];
#pragma unroll
    for (int r = 0; r < 8; ++r) {
      int m = msub * 16 + r + 8 * half;
      size_t grow = (size_t)(rowbase + m);
      float iv = sigm(ai[r] + bi_);
      float fv = sigm(af[r] + bf_);
      float gv = fast_tanh(ag[r] + bg_);
      float ov = sigm(ao[r] + bo_);
      float c0v = c0[grow * 256 + col];
      float cn = fv * c0v + iv * gv;
      float hn = ov * fast_tanh(cn);
      outC[grow * 256 + col] = cn;
      outH[grow * 256 + col] = hn;
      hbuf[m * 256 + col] = hn;
    }
  }
  __syncthreads();

  // ---- stage 7: LN(h_new) -> Aq1 (f16) ----
  ln_rows(hbuf, 256, ln_g, ln_b, Aq1, 256, false, tid);
  __syncthreads();

  // ---- stage 8: q1 GEMM (32x256 @ 256x128) + bias -> f32 ----
  for (int u = wave; u < 16; u += 8) {
    int msub = u >> 3, nt = u & 7;
    v8f acc = gemm_acc(Aq1, 256, (const char*)ws + OFF_WQ1, 8, msub, nt, lane);
    int n = nt * 16 + Nl; float bv = q1_bv[n];
#pragma unroll
    for (int r = 0; r < 8; ++r) q1o[(msub * 16 + r + 8 * half) * 128 + n] = acc[r] + bv;
  }
  __syncthreads();

  // ---- stage 9: LN(q1)+relu -> Aq2 (f16) ----
  ln_rows(q1o, 128, q1_g, q1_beta, Aq2, 128, true, tid);
  __syncthreads();

  // ---- stage 10: q2 head (32x128 @ 128x16), cols 0..7 valid ----
  for (int u = wave; u < 2; u += 8) {
    int msub = u;
    v8f acc = gemm_acc(Aq2, 128, (const char*)ws + OFF_WQ2, 4, msub, 0, lane);
    if (Nl < 8) {
      float bv = bq2[Nl];
#pragma unroll
      for (int r = 0; r < 8; ++r) {
        size_t grow = (size_t)(rowbase + msub * 16 + r + 8 * half);
        outQ[grow * 8 + Nl] = acc[r] + bv;
      }
    }
  }
}

extern "C" void kernel_launch(void* const* d_in, const int* in_sizes, int n_in,
                              void* d_out, int out_size, void* d_ws, size_t ws_size,
                              hipStream_t stream) {
  (void)n_in; (void)out_size; (void)ws_size;
  const float* obs      = (const float*)d_in[0];
  const float* h0       = (const float*)d_in[1];
  const float* c0       = (const float*)d_in[2];
  const float* conv1_w  = (const float*)d_in[3];
  const float* conv1_b  = (const float*)d_in[4];
  const float* conv2_w  = (const float*)d_in[5];
  const float* conv2_b  = (const float*)d_in[6];
  const float* vis_w    = (const float*)d_in[7];
  const float* vis_b    = (const float*)d_in[8];
  const float* vis_g    = (const float*)d_in[9];
  const float* vis_beta = (const float*)d_in[10];
  const float* pos_w    = (const float*)d_in[11];
  const float* pos_b    = (const float*)d_in[12];
  const float* met_w    = (const float*)d_in[13];
  const float* met_b    = (const float*)d_in[14];
  const float* aff_w    = (const float*)d_in[15];
  const float* aff_b    = (const float*)d_in[16];
  const float* w_ih     = (const float*)d_in[17];
  const float* w_hh     = (const float*)d_in[18];
  const float* b_ih     = (const float*)d_in[19];
  const float* b_hh     = (const float*)d_in[20];
  const float* ln_g     = (const float*)d_in[21];
  const float* ln_b     = (const float*)d_in[22];
  const float* q1_w     = (const float*)d_in[23];
  const float* q1_b     = (const float*)d_in[24];
  const float* q1_g     = (const float*)d_in[25];
  const float* q1_beta  = (const float*)d_in[26];
  const float* q2_w     = (const float*)d_in[27];
  const float* q2_b     = (const float*)d_in[28];
  char* ws = (char*)d_ws;
  int Btot = in_sizes[0] / 50;

  prep_weights<<<(PREP_TOTAL + 255) / 256, 256, 0, stream>>>(
      conv1_w, conv1_b, conv2_w, conv2_b, vis_w,
      pos_w, pos_b, met_w, met_b, aff_w, aff_b,
      w_ih, w_hh, b_ih, b_hh, q1_w, q2_w, q2_b, ws);

  rsqn_fused_wmma<<<Btot / 32, 256, SMEM_BYTES, stream>>>(
      obs, h0, c0, vis_b, vis_g, vis_beta, ln_g, ln_b,
      q1_b, q1_g, q1_beta, ws, (float*)d_out, Btot);
}
